// TransformerModel_52132313039285
// MI455X (gfx1250) — matrix operand order repaired and checked by reading, not compiled, not run on hardware
//
#include <hip/hip_runtime.h>
#include <cstdint>
#include <cstddef>

// ---------------------------------------------------------------------------
// Transformer forward on MI455X (gfx1250).
//  - All GEMMs + both attention matmuls on v_wmma_f32_16x16x32_bf16.
//  - GEMM K-tiles staged by the Tensor Data Mover (TENSOR_LOAD_TO_LDS,
//    TENSORcnt), triple-buffered -> one barrier per K tile, DMA overlaps WMMA.
//    (falls back to GLOBAL_LOAD_ASYNC_TO_LDS_B128 if the builtin is absent)
//  - Attention K staging via async global->LDS (ASYNCcnt).
//  - Softmax reductions on v_permlane16_b32 (VALU) instead of ds_bpermute.
// Workspace need ~140 MB.
// ---------------------------------------------------------------------------

#define TV  32000
#define TD  1024
#define THH 16
#define THD 64
#define TNL 6
#define TL  1024
#define TBB 2
#define TM  (TBB * TL)   // 2048 token rows

typedef __attribute__((ext_vector_type(16))) __bf16 v16bf;
typedef __attribute__((ext_vector_type(8)))  float  v8f;
typedef __attribute__((ext_vector_type(4)))  unsigned int u32x4;
typedef __attribute__((ext_vector_type(8)))  int i32x8;
typedef __attribute__((ext_vector_type(4)))  int i32x4;

struct alignas(16) U4 { unsigned int x, y, z, w; };
union FragAB { v16bf v; U4 u[2]; };

__device__ __forceinline__ unsigned short f2bf(float f) {
  unsigned int u = __float_as_uint(f);
  u += 0x7FFFu + ((u >> 16) & 1u);          // round-to-nearest-even
  return (unsigned short)(u >> 16);
}

__device__ __forceinline__ v8f wmma_bf16(FragAB a, FragAB b, v8f c) {
  return __builtin_amdgcn_wmma_f32_16x16x32_bf16(
      /*neg_a=*/false, a.v, /*neg_b=*/false, b.v,
      /*c_mod=*/(short)0, c, /*reuse_a=*/false, /*reuse_b=*/false);
}

// --- CDNA5 async global->LDS copy (ASYNCcnt) -------------------------------
__device__ __forceinline__ void async_ld_b128(unsigned lds_off, const void* g) {
  unsigned long long ga = (unsigned long long)g;
  asm volatile("global_load_async_to_lds_b128 %0, %1, off"
               :: "v"(lds_off), "v"(ga) : "memory");
}
__device__ __forceinline__ void wait_async_le4() {
  asm volatile("s_wait_asynccnt 0x4" ::: "memory");
}
__device__ __forceinline__ void wait_async_0() {
  asm volatile("s_wait_asynccnt 0x0" ::: "memory");
}
__device__ __forceinline__ unsigned lds_off32(const void* p) {
  return (unsigned)(unsigned long long)p;   // low 32 bits of LDS aperture addr
}

// --- Tensor Data Mover: one D# per 2-D tile (CDNA5 ISA) ----
#if __has_builtin(__builtin_amdgcn_tensor_load_to_lds)
#define HAVE_TDM 1
#else
#define HAVE_TDM 0
#endif

#if HAVE_TDM
// rows x row_dw DWORD tile, global row stride = stride_dw DWORDs.
// LDS rows padded: pad_interval=3 (16 DW = 64B row), pad_amount=3 (4 DW = 16B)
// -> reproduces the 40-half padded LDS rows used by the compute fragments.
__device__ __forceinline__ void tdm_load_tile(const void* gsrc, unsigned lds_byte,
                                              unsigned rows, unsigned row_dw,
                                              unsigned stride_dw) {
  const unsigned long long ga = (unsigned long long)gsrc;
  u32x4 g0;
  g0[0] = 1u;                                                   // count=1
  g0[1] = lds_byte;                                             // lds_addr
  g0[2] = (unsigned)ga;                                         // addr[31:0]
  g0[3] = (unsigned)((ga >> 32) & 0x01FFFFFFull) | (2u << 30);  // addr[56:32], type=2
  i32x8 g1;
  g1[0] = (int)((2u << 16) | (1u << 20) | (3u << 22) | (3u << 25)); // 4B elems, LDS pad
  g1[1] = (int)((row_dw & 0xFFFFu) << 16);                      // tensor_dim0[15:0]
  g1[2] = (int)(((row_dw >> 16) & 0xFFFFu) | ((rows & 0xFFFFu) << 16)); // dim0 hi, dim1 lo
  g1[3] = (int)(((rows >> 16) & 0xFFFFu) | ((row_dw & 0xFFFFu) << 16)); // dim1 hi, tile_dim0
  g1[4] = (int)(rows & 0xFFFFu);                                // tile_dim1 (tile_dim2=0)
  g1[5] = (int)stride_dw;                                       // tensor_dim0_stride[31:0]
  g1[6] = 0;
  g1[7] = 0;
  const i32x4 z = {0, 0, 0, 0};
#if __clang_major__ >= 23
  const i32x8 z8 = {0, 0, 0, 0, 0, 0, 0, 0};
  __builtin_amdgcn_tensor_load_to_lds(g0, g1, z, z, z8, 0);
#else
  __builtin_amdgcn_tensor_load_to_lds(g0, g1, z, z, 0);
#endif
}
#endif

// --- 16-lane butterfly reductions via v_permlane16_b32 (VALU, no LDS) ------
__device__ __forceinline__ float pl16x(float v, unsigned lo, unsigned hi) {
#if __has_builtin(__builtin_amdgcn_permlane16)
  return __uint_as_float(__builtin_amdgcn_permlane16(
      __float_as_uint(v), __float_as_uint(v), lo, hi, false, false));
#else
  (void)lo; (void)hi;
  return v;
#endif
}
__device__ __forceinline__ float rowmax16(float v) {
#if __has_builtin(__builtin_amdgcn_permlane16)
  v = fmaxf(v, pl16x(v, 0x67452301u, 0xEFCDAB89u));   // lane ^ 1
  v = fmaxf(v, pl16x(v, 0x54761032u, 0xDCFE98BAu));   // lane ^ 2
  v = fmaxf(v, pl16x(v, 0x32107654u, 0xBA98FEDCu));   // lane ^ 4
  v = fmaxf(v, pl16x(v, 0xFEDCBA98u, 0x76543210u));   // lane ^ 8
#else
  v = fmaxf(v, __shfl_xor(v, 1, 32));
  v = fmaxf(v, __shfl_xor(v, 2, 32));
  v = fmaxf(v, __shfl_xor(v, 4, 32));
  v = fmaxf(v, __shfl_xor(v, 8, 32));
#endif
  return v;
}
__device__ __forceinline__ float rowsum16(float v) {
#if __has_builtin(__builtin_amdgcn_permlane16)
  v += pl16x(v, 0x67452301u, 0xEFCDAB89u);
  v += pl16x(v, 0x54761032u, 0xDCFE98BAu);
  v += pl16x(v, 0x32107654u, 0xBA98FEDCu);
  v += pl16x(v, 0xFEDCBA98u, 0x76543210u);
#else
  v += __shfl_xor(v, 1, 32);
  v += __shfl_xor(v, 2, 32);
  v += __shfl_xor(v, 4, 32);
  v += __shfl_xor(v, 8, 32);
#endif
  return v;
}

// ---------------------------------------------------------------------------
// x = emb[tokens] + pos
// ---------------------------------------------------------------------------
__global__ void embed_kernel(const int* __restrict__ tokens,
                             const float* __restrict__ emb,
                             const float* __restrict__ pos,
                             float* __restrict__ x) {
  const int row = blockIdx.x;
  const int l   = row % TL;
  const int tok = tokens[row];
  const float* e = emb + (size_t)tok * TD;
  const float* p = pos + (size_t)l * TD;
  float* xr = x + (size_t)row * TD;
  for (int i = threadIdx.x; i < TD; i += blockDim.x) xr[i] = e[i] + p[i];
}

// ---------------------------------------------------------------------------
// LayerNorm over D=1024, one block (256 thr) per row, bf16 output
// ---------------------------------------------------------------------------
__global__ __launch_bounds__(256)
void ln_kernel(const float* __restrict__ x, const float* __restrict__ g,
               const float* __restrict__ bb, unsigned short* __restrict__ out) {
  __shared__ float red[256];
  const int row = blockIdx.x, tid = threadIdx.x;
  const float* xr = x + (size_t)row * TD;
  float v[4];
  float s = 0.f;
#pragma unroll
  for (int i = 0; i < 4; i++) { v[i] = xr[tid + i * 256]; s += v[i]; }
  red[tid] = s; __syncthreads();
  for (int o = 128; o > 0; o >>= 1) { if (tid < o) red[tid] += red[tid + o]; __syncthreads(); }
  const float mu = red[0] * (1.0f / TD);
  __syncthreads();
  s = 0.f;
#pragma unroll
  for (int i = 0; i < 4; i++) { float d = v[i] - mu; s += d * d; }
  red[tid] = s; __syncthreads();
  for (int o = 128; o > 0; o >>= 1) { if (tid < o) red[tid] += red[tid + o]; __syncthreads(); }
  const float rstd = rsqrtf(red[0] * (1.0f / TD) + 1e-5f);
  unsigned short* orow = out + (size_t)row * TD;
#pragma unroll
  for (int i = 0; i < 4; i++) {
    int c = tid + i * 256;
    orow[c] = f2bf((v[i] - mu) * rstd * g[c] + bb[c]);
  }
}

// ---------------------------------------------------------------------------
// fp32 [K,N] -> bf16 transposed [N,K] (LDS-tiled for coalescing)
// ---------------------------------------------------------------------------
__global__ void conv_transpose_bf16(const float* __restrict__ W,
                                    unsigned short* __restrict__ Wt,
                                    int K, int N) {
  __shared__ float tile[32][33];
  const int n0 = blockIdx.x * 32, k0 = blockIdx.y * 32;
  const int tx = threadIdx.x, ty = threadIdx.y;
#pragma unroll
  for (int i = 0; i < 4; i++) {
    int k = k0 + ty + i * 8;
    tile[ty + i * 8][tx] = W[(size_t)k * N + n0 + tx];
  }
  __syncthreads();
#pragma unroll
  for (int i = 0; i < 4; i++) {
    int n = n0 + ty + i * 8;
    Wt[(size_t)n * K + k0 + tx] = f2bf(tile[tx][ty + i * 8]);
  }
}

__global__ void f32_to_bf16_kernel(const float* __restrict__ in,
                                   unsigned short* __restrict__ out, int n) {
  int i = blockIdx.x * blockDim.x + threadIdx.x;
  if (i < n) out[i] = f2bf(in[i]);
}

// ---------------------------------------------------------------------------
// bf16 WMMA GEMM: C[M,N] = A[M,K] * Bt[N,K]^T (+ epilogue)
//   EPI 0: Cb = bf16(acc + bias)
//   EPI 1: Cf += acc + bias              (residual add into fp32 stream)
//   EPI 2: Cf += gelu(acc + bias)        (reference applies GELU after W2)
//   EPI 3: Cf  = acc                     (logits, no bias)
// 128x128 macro tile, 8 waves (2Mx4N), 64x32 per wave.
// K staged 32-wide by the TDM (2 tensor loads / tile, issued by wave 0),
// triple-buffered, 1 barrier per tile; DMA for tile k+1 overlaps WMMA of k.
// ---------------------------------------------------------------------------
template <int EPI>
__global__ __launch_bounds__(256)
void gemm_bf16_wmma(const unsigned short* __restrict__ A,
                    const unsigned short* __restrict__ Bt,
                    const float* __restrict__ bias,
                    float* __restrict__ Cf,
                    unsigned short* __restrict__ Cb,
                    int M, int N, int K) {
  __shared__ unsigned short As[3][128][40];   // 3 buffers, padded rows (80B)
  __shared__ unsigned short Bs[3][128][40];
  const int tid   = threadIdx.x;
  const int lane  = tid & 31;
  const int wid   = tid >> 5;
  const int waveM = wid >> 2;
  const int waveN = wid & 3;
  const int lr    = lane & 15;
  const int hi    = (lane >> 4) & 1;
  const int m0    = blockIdx.y * 128;
  const int n0    = blockIdx.x * 128;

  const v8f zero = {0.f, 0.f, 0.f, 0.f, 0.f, 0.f, 0.f, 0.f};
  v8f acc[4][2];
#pragma unroll
  for (int i = 0; i < 4; i++)
#pragma unroll
    for (int j = 0; j < 2; j++) acc[i][j] = zero;

#if HAVE_TDM
  // wave-scalar DMA: whole 128x32-half tile per tensor op (EXEC-independent)
  auto issue = [&](int kt, int buf) {
    tdm_load_tile(A  + (size_t)m0 * K + kt * 32, lds_off32(&As[buf][0][0]),
                  128u, 16u, (unsigned)(K / 2));
    tdm_load_tile(Bt + (size_t)n0 * K + kt * 32, lds_off32(&Bs[buf][0][0]),
                  128u, 16u, (unsigned)(K / 2));
  };
  const int nk = K / 32;
  if (wid == 0) issue(0, 0);
  for (int kt = 0; kt < nk; ++kt) {
    const int cur = kt % 3;
    if (wid == 0) {
      if (kt + 1 < nk) {
        issue(kt + 1, (kt + 1) % 3);
        __builtin_amdgcn_s_wait_tensorcnt(2);   // oldest 2 (= tile kt) complete
      } else {
        __builtin_amdgcn_s_wait_tensorcnt(0);
      }
    }
    __syncthreads();                  // tile kt visible to all waves
#else
  const int rowS = tid >> 1;          // 0..127
  const int cb   = (tid & 1) * 16;    // half-element column base (0 or 16)
  auto issue = [&](int kt, int buf) {
    const unsigned short* ga = A  + (size_t)(m0 + rowS) * K + kt * 32 + cb;
    const unsigned short* gb = Bt + (size_t)(n0 + rowS) * K + kt * 32 + cb;
    const unsigned la = lds_off32(&As[buf][rowS][cb]);
    const unsigned lb = lds_off32(&Bs[buf][rowS][cb]);
    async_ld_b128(la,      ga);
    async_ld_b128(la + 16, ga + 8);
    async_ld_b128(lb,      gb);
    async_ld_b128(lb + 16, gb + 8);
  };
  const int nk = K / 32;
  issue(0, 0);
  for (int kt = 0; kt < nk; ++kt) {
    const int cur = kt % 3;
    if (kt + 1 < nk) { issue(kt + 1, (kt + 1) % 3); wait_async_le4(); }
    else             { wait_async_0(); }
    __syncthreads();
#endif

    FragAB a[4], b[2];
#pragma unroll
    for (int sm = 0; sm < 4; sm++) {  // A frag: lane<16 K{0..7,16..23}, hi K{8..15,24..31}
      const unsigned short* p = &As[cur][waveM * 64 + sm * 16 + lr][hi * 8];
      a[sm].u[0] = *(const U4*)p;
      a[sm].u[1] = *(const U4*)(p + 16);
    }
#pragma unroll
    for (int sn = 0; sn < 2; sn++) {  // B frag: lane<16 K{0..7,16..23}, hi K{8..15,24..31}
      const unsigned short* p = &Bs[cur][waveN * 32 + sn * 16 + lr][hi * 8];
      b[sn].u[0] = *(const U4*)p;
      b[sn].u[1] = *(const U4*)(p + 16);
    }
#pragma unroll
    for (int sm = 0; sm < 4; sm++)
#pragma unroll
      for (int sn = 0; sn < 2; sn++)
        acc[sm][sn] = wmma_bf16(a[sm], b[sn], acc[sm][sn]);
  }

  // Epilogue: C layout -> (row = r + hi*8, col = lr) per 16x16 tile
#pragma unroll
  for (int sm = 0; sm < 4; sm++) {
#pragma unroll
    for (int sn = 0; sn < 2; sn++) {
#pragma unroll
      for (int r = 0; r < 8; r++) {
        const int row = m0 + waveM * 64 + sm * 16 + r + hi * 8;
        const int col = n0 + waveN * 32 + sn * 16 + lr;
        const float v = acc[sm][sn][r];
        if (EPI == 0) {
          Cb[(size_t)row * N + col] = f2bf(v + bias[col]);
        } else if (EPI == 1) {
          Cf[(size_t)row * N + col] += v + bias[col];
        } else if (EPI == 2) {
          float t = v + bias[col];
          Cf[(size_t)row * N + col] += 0.5f * t * (1.0f + erff(t * 0.70710678118654752f));
        } else {
          Cf[(size_t)row * N + col] = v;
        }
      }
    }
  }
}

// ---------------------------------------------------------------------------
// Flash attention: one block = (b, h, 64-row query tile); 4 waves x 16 rows.
// S = Q K^T and O += P V both via WMMA bf16. K rows staged with async
// global->LDS; online softmax reductions via v_permlane16_b32.
// ---------------------------------------------------------------------------
__global__ __launch_bounds__(128)
void attn_kernel(const unsigned short* __restrict__ qkv,  // [TM, 3*TD] bf16
                 unsigned short* __restrict__ y) {        // [TM, TD]  bf16
  __shared__ unsigned short Kt[32][72];       // key rows, 64 feats (+pad)
  __shared__ unsigned short Vt[64][40];       // V transposed: [feat][key]
  __shared__ unsigned short Pl[4][16][40];    // per-wave P relayout buffer

  const int qb   = blockIdx.x * 64;
  const int h    = blockIdx.y;
  const int b    = blockIdx.z;
  const int tid  = threadIdx.x;
  const int lane = tid & 31, wave = tid >> 5;
  const int lr   = lane & 15, hi = (lane >> 4) & 1;

  const unsigned short* qbase = qkv + (size_t)b * TL * (3 * TD) + h * THD;
  const unsigned short* kbase = qbase + TD;
  const unsigned short* vbase = qbase + 2 * TD;

  // Q as A-fragments (two K-steps of 32 covering HD=64), straight from global
  FragAB q0, q1;
  {
    const unsigned short* qr = qbase + (size_t)(qb + wave * 16 + lr) * (3 * TD);
    q0.u[0] = *(const U4*)(qr + hi * 8);
    q0.u[1] = *(const U4*)(qr + hi * 8 + 16);
    q1.u[0] = *(const U4*)(qr + 32 + hi * 8);
    q1.u[1] = *(const U4*)(qr + 32 + hi * 8 + 16);
  }

  const v8f zero = {0.f, 0.f, 0.f, 0.f, 0.f, 0.f, 0.f, 0.f};
  v8f o[4];
#pragma unroll
  for (int f = 0; f < 4; f++) o[f] = zero;
  float mrow[8], lrow[8];
#pragma unroll
  for (int r = 0; r < 8; r++) { mrow[r] = -1.0e30f; lrow[r] = 0.f; }

  for (int j0 = 0; j0 <= qb + 32; j0 += 32) {   // causal: keys up to qb+63
    __syncthreads();
    {   // stage 32 key rows (async->LDS) and V transposed (manual) into LDS
      const int r = tid >> 3, c = (tid & 7) * 8;
      async_ld_b128(lds_off32(&Kt[r][c]),
                    kbase + (size_t)(j0 + r) * (3 * TD) + c);
      async_ld_b128(lds_off32(&Kt[r + 16][c]),
                    kbase + (size_t)(j0 + r + 16) * (3 * TD) + c);
#pragma unroll
      for (int p = 0; p < 2; p++) {
        const int id = tid + p * 128;
        const int key = id >> 3, ch = (id & 7) * 8;
        U4 d = *(const U4*)(vbase + (size_t)(j0 + key) * (3 * TD) + ch);
        const unsigned short* hv = (const unsigned short*)&d;
#pragma unroll
        for (int e = 0; e < 8; e++) Vt[ch + e][key] = hv[e];
      }
      wait_async_0();
    }
    __syncthreads();

    // S = Q K^T for two 16-key subtiles
    v8f st[2];
#pragma unroll
    for (int sn = 0; sn < 2; sn++) {
      FragAB kb0, kb1;
      const unsigned short* kp = &Kt[sn * 16 + lr][hi * 8];
      kb0.u[0] = *(const U4*)kp;        kb0.u[1] = *(const U4*)(kp + 16);
      kb1.u[0] = *(const U4*)(kp + 32); kb1.u[1] = *(const U4*)(kp + 48);
      v8f s = zero;
      s = wmma_bf16(q0, kb0, s);
      s = wmma_bf16(q1, kb1, s);
#pragma unroll
      for (int r = 0; r < 8; r++) {
        const int rowq = qb + wave * 16 + r + hi * 8;
        const int kc   = j0 + sn * 16 + lr;
        s[r] = (kc <= rowq) ? s[r] * 0.125f : -1.0e30f;   // 1/sqrt(64)
      }
      st[sn] = s;
    }

    // online softmax; butterfly reductions stay inside 16-lane halves
#pragma unroll
    for (int r = 0; r < 8; r++) {
      const float mx    = rowmax16(fmaxf(st[0][r], st[1][r]));
      const float mn    = fmaxf(mrow[r], mx);
      const float alpha = __expf(mrow[r] - mn);
      mrow[r] = mn;
      const float p0 = __expf(st[0][r] - mn);
      const float p1 = __expf(st[1][r] - mn);
      lrow[r] = lrow[r] * alpha + rowsum16(p0 + p1);
      Pl[wave][r + hi * 8][lr]      = f2bf(p0);
      Pl[wave][r + hi * 8][16 + lr] = f2bf(p1);
#pragma unroll
      for (int f = 0; f < 4; f++) o[f][r] *= alpha;
    }

    // O += P V  (P re-read from LDS in A-fragment layout; same-wave DS order)
    FragAB pa;
    const unsigned short* pp = &Pl[wave][lr][hi * 8];
    pa.u[0] = *(const U4*)pp;
    pa.u[1] = *(const U4*)(pp + 16);
#pragma unroll
    for (int f = 0; f < 4; f++) {
      FragAB vb;
      const unsigned short* vp = &Vt[f * 16 + lr][hi * 8];
      vb.u[0] = *(const U4*)vp;
      vb.u[1] = *(const U4*)(vp + 16);
      o[f] = wmma_bf16(pa, vb, o[f]);
    }
  }

  // normalize and write bf16 output at [row, h*64 + feat]
  unsigned short* yr = y + (size_t)(b * TL + qb + wave * 16) * TD + h * THD;
#pragma unroll
  for (int r = 0; r < 8; r++) {
    const float inv = 1.0f / lrow[r];
#pragma unroll
    for (int f = 0; f < 4; f++)
      yr[(size_t)(r + hi * 8) * TD + f * 16 + lr] = f2bf(o[f][r] * inv);
  }
}

// ---------------------------------------------------------------------------
extern "C" void kernel_launch(void* const* d_in, const int* in_sizes, int n_in,
                              void* d_out, int out_size, void* d_ws, size_t ws_size,
                              hipStream_t stream) {
  (void)in_sizes; (void)n_in; (void)out_size; (void)ws_size;
  const int*   tokens = (const int*)  d_in[0];
  const float* pos    = (const float*)d_in[1];
  const float* emb    = (const float*)d_in[2];
  const float* ln1_g  = (const float*)d_in[3];
  const float* ln1_b  = (const float*)d_in[4];
  const float* ln2_g  = (const float*)d_in[5];
  const float* ln2_b  = (const float*)d_in[6];
  const float* attn_w = (const float*)d_in[7];
  const float* attn_b = (const float*)d_in[8];
  const float* proj_w = (const float*)d_in[9];
  const float* proj_b = (const float*)d_in[10];
  const float* mlp_w1 = (const float*)d_in[11];
  const float* mlp_b1 = (const float*)d_in[12];
  const float* mlp_w2 = (const float*)d_in[13];
  const float* mlp_b2 = (const float*)d_in[14];
  const float* out_w  = (const float*)d_in[15];
  float* logits = (float*)d_out;

  char* ws = (char*)d_ws;
  size_t off = 0;
  auto alloc = [&](size_t bytes) -> char* {
    char* p = ws + off;
    off = (off + bytes + 255) & ~(size_t)255;
    return p;
  };
  float*          xf   = (float*)         alloc((size_t)TM * TD * 4);
  unsigned short* hln  = (unsigned short*)alloc((size_t)TM * TD * 2);
  unsigned short* qkvb = (unsigned short*)alloc((size_t)TM * 3 * TD * 2);
  unsigned short* atny = (unsigned short*)alloc((size_t)TM * TD * 2);
  unsigned short* mid  = (unsigned short*)alloc((size_t)TM * 4 * TD * 2);
  unsigned short* xb   = (unsigned short*)alloc((size_t)TM * TD * 2);
  unsigned short* wA   = (unsigned short*)alloc((size_t)3 * TD * TD * 2);
  unsigned short* wP   = (unsigned short*)alloc((size_t)TD * TD * 2);
  unsigned short* w1t  = (unsigned short*)alloc((size_t)4 * TD * TD * 2);
  unsigned short* w2t  = (unsigned short*)alloc((size_t)4 * TD * TD * 2);
  unsigned short* wOt  = (unsigned short*)alloc((size_t)TV * TD * 2);

  const dim3 tb(32, 8);
  embed_kernel<<<TM, 256, 0, stream>>>(tokens, emb, pos, xf);

  for (int i = 0; i < TNL; ++i) {
    conv_transpose_bf16<<<dim3(3 * TD / 32, TD / 32), tb, 0, stream>>>(
        attn_w + (size_t)i * TD * 3 * TD, wA, TD, 3 * TD);
    ln_kernel<<<TM, 256, 0, stream>>>(xf, ln1_g + i * TD, ln1_b + i * TD, hln);
    gemm_bf16_wmma<0><<<dim3(3 * TD / 128, TM / 128), 256, 0, stream>>>(
        hln, wA, attn_b + (size_t)i * 3 * TD, nullptr, qkvb, TM, 3 * TD, TD);
    attn_kernel<<<dim3(TL / 64, THH, TBB), 128, 0, stream>>>(qkvb, atny);
    conv_transpose_bf16<<<dim3(TD / 32, TD / 32), tb, 0, stream>>>(
        proj_w + (size_t)i * TD * TD, wP, TD, TD);
    gemm_bf16_wmma<1><<<dim3(TD / 128, TM / 128), 256, 0, stream>>>(
        atny, wP, proj_b + i * TD, xf, nullptr, TM, TD, TD);
    ln_kernel<<<TM, 256, 0, stream>>>(xf, ln2_g + i * TD, ln2_b + i * TD, hln);
    conv_transpose_bf16<<<dim3(4 * TD / 32, TD / 32), tb, 0, stream>>>(
        mlp_w1 + (size_t)i * TD * 4 * TD, w1t, TD, 4 * TD);
    gemm_bf16_wmma<0><<<dim3(4 * TD / 128, TM / 128), 256, 0, stream>>>(
        hln, w1t, mlp_b1 + (size_t)i * 4 * TD, nullptr, mid, TM, 4 * TD, TD);
    conv_transpose_bf16<<<dim3(TD / 32, 4 * TD / 32), tb, 0, stream>>>(
        mlp_w2 + (size_t)i * 4 * TD * TD, w2t, 4 * TD, TD);
    gemm_bf16_wmma<2><<<dim3(TD / 128, TM / 128), 256, 0, stream>>>(
        mid, w2t, mlp_b2 + i * TD, xf, nullptr, TM, TD, 4 * TD);
  }

  f32_to_bf16_kernel<<<(TM * TD + 255) / 256, 256, 0, stream>>>(xf, xb, TM * TD);
  conv_transpose_bf16<<<dim3(TV / 32, TD / 32), tb, 0, stream>>>(out_w, wOt, TD, TV);
  gemm_bf16_wmma<3><<<dim3(TV / 128, TM / 128), 256, 0, stream>>>(
      xb, wOt, nullptr, logits, nullptr, TM, TV, TD);
}
